// AttentionPooling_10445360464522
// MI455X (gfx1250) — matrix operand order repaired and checked by reading, not compiled, not run on hardware
//
#include <hip/hip_runtime.h>
#include <stdint.h>

#define D_DIM 256
#define H_DIM 128
#define NSEG  4096
#define ROWS_PER_BLOCK 128
#define XPITCH 258   // ushorts per x-tile row (256 + 2 pad -> conflict-free LDS)
#define WPITCH 258   // ushorts per W1^T column (256 + 2 pad)

typedef __attribute__((ext_vector_type(16))) __bf16 v16bf;
typedef __attribute__((ext_vector_type(8)))  float  v8f;

union BfFrag {
  unsigned short u16[16];
  unsigned int   u32[8];
  v16bf          v;
};

__device__ __forceinline__ unsigned short f32_to_bf16_rne(float f) {
  unsigned int b = __float_as_uint(f);
  b += 0x7FFFu + ((b >> 16) & 1u);   // round-to-nearest-even
  return (unsigned short)(b >> 16);
}

// CDNA5 hardware tanh (TRANS op). v_nop satisfies the 1-op TRANS result hazard.
__device__ __forceinline__ float tanh_hw(float x) {
  float r;
  asm volatile("v_tanh_f32 %0, %1\n\tv_nop" : "=v"(r) : "v"(x));
  return r;
}

// ---------------------------------------------------------------------------
// Kernel 1: scores[n] = tanh(x@W1 + b1) @ W2 + b2   via bf16 WMMA
// Block = 256 threads (8 waves), 128 rows per block; wave w owns rows w*16..+15
// ---------------------------------------------------------------------------
__global__ __launch_bounds__(256) void attn_scores_kernel(
    const float* __restrict__ x, const float* __restrict__ W1,
    const float* __restrict__ b1, const float* __restrict__ W2,
    const float* __restrict__ b2, float* __restrict__ scores, int n)
{
  __shared__ unsigned short s_w1t[H_DIM * WPITCH];        // W1^T in bf16: [col][k]
  __shared__ unsigned short s_x[ROWS_PER_BLOCK * XPITCH]; // x tile bf16: [row][k]

  const int tid = threadIdx.x;
  const int r0  = blockIdx.x * ROWS_PER_BLOCK;

  // Stage W1 transposed (global-coalesced read, conflict-free LDS write)
  for (int i = tid; i < D_DIM * H_DIM; i += 256) {
    int col = i & (H_DIM - 1);
    int k   = i >> 7;
    s_w1t[col * WPITCH + k] = f32_to_bf16_rne(W1[(size_t)k * H_DIM + col]);
  }
  // Stage x tile, zero-pad tail rows
  for (int i = tid; i < ROWS_PER_BLOCK * (D_DIM / 4); i += 256) {
    int row = i >> 6;
    int c4  = (i & 63) << 2;
    int r   = r0 + row;
    float4 val = make_float4(0.f, 0.f, 0.f, 0.f);
    if (r < n) val = *(const float4*)(x + (size_t)r * D_DIM + c4);
    unsigned short* dst = &s_x[row * XPITCH + c4];
    dst[0] = f32_to_bf16_rne(val.x);
    dst[1] = f32_to_bf16_rne(val.y);
    dst[2] = f32_to_bf16_rne(val.z);
    dst[3] = f32_to_bf16_rne(val.w);
  }
  __syncthreads();

  const int wave = tid >> 5;
  const int lane = tid & 31;
  const int lm   = lane & 15;
  const int hi   = lane >> 4;

  v8f acc[8] = {};   // 8 column tiles of 16x16 f32 accumulators (h before tanh)

  const unsigned short* xrow = &s_x[(wave * 16 + lm) * XPITCH];

  for (int kb = 0; kb < 8; ++kb) {       // K = 256 in steps of 32
    const int kbase = kb * 32;
    // A fragment: 16-bit A 16x32 layout (pairs contiguous -> ds_load_b32)
    BfFrag a;
#pragma unroll
    for (int v = 0; v < 4; ++v) {
      int k = kbase + 2 * v + hi * 16;
      a.u32[v] = *(const unsigned int*)(xrow + k);
    }
#pragma unroll
    for (int v = 4; v < 8; ++v) {
      int k = kbase + 8 + 2 * (v - 4) + hi * 16;
      a.u32[v] = *(const unsigned int*)(xrow + k);
    }
#pragma unroll
    for (int c = 0; c < 8; ++c) {
      // B fragment: 16-bit B 32x16 layout; W1^T makes (k,k+1) pairs contiguous
      const unsigned short* wcol = &s_w1t[(c * 16 + lm) * WPITCH];
      BfFrag bfr;
#pragma unroll
      for (int v = 0; v < 8; ++v) {
        int k = kbase + 2 * v + hi * 16;
        bfr.u32[v] = *(const unsigned int*)(wcol + k);
      }
      acc[c] = __builtin_amdgcn_wmma_f32_16x16x32_bf16(
          false, a.v, false, bfr.v, (short)0, acc[c], false, false);
    }
  }

  // Epilogue: C/D layout => lane holds col n=lm, rows m = i + 8*hi (i=0..7)
  float p[8];
#pragma unroll
  for (int i = 0; i < 8; ++i) p[i] = 0.f;
#pragma unroll
  for (int c = 0; c < 8; ++c) {
    int col   = c * 16 + lm;
    float b1v = b1[col];
    float w2v = W2[col];
#pragma unroll
    for (int i = 0; i < 8; ++i)
      p[i] += tanh_hw(acc[c][i] + b1v) * w2v;
  }
  // reduce over the 16 columns held by each half-wave
#pragma unroll
  for (int i = 0; i < 8; ++i) {
    float s = p[i];
    s += __shfl_xor(s, 1, 32);
    s += __shfl_xor(s, 2, 32);
    s += __shfl_xor(s, 4, 32);
    s += __shfl_xor(s, 8, 32);
    p[i] = s;
  }
  if (lm == 0) {
    float b2v = b2[0];
#pragma unroll
    for (int i = 0; i < 8; ++i) {
      int r = r0 + wave * 16 + hi * 8 + i;
      if (r < n) scores[r] = p[i] + b2v;
    }
  }
}

// ---------------------------------------------------------------------------
// Kernel 2: segment boundaries from the sorted batch array (int64)
// seg_start[g] = first node of segment g; seg_start[NSEG] = n (empty segs ok)
// ---------------------------------------------------------------------------
__global__ void seg_bounds_kernel(const long long* __restrict__ batch,
                                  int* __restrict__ seg_start, int n)
{
  int i = blockIdx.x * blockDim.x + threadIdx.x;
  if (i >= n) return;
  int cur  = (int)batch[i];
  int prev = (i == 0) ? -1 : (int)batch[i - 1];
  for (int s = prev + 1; s <= cur; ++s) seg_start[s] = i;
  if (i == n - 1)
    for (int s = cur + 1; s <= NSEG; ++s) seg_start[s] = n;
}

// ---------------------------------------------------------------------------
// Kernel 3: per-segment stable softmax stats (deterministic, no atomics)
// ex[i] = exp(score - segmax); seg_recip[g] = 1 / (sum(ex) + 1e-16)
// ---------------------------------------------------------------------------
__global__ __launch_bounds__(256) void seg_stats_kernel(
    const float* __restrict__ scores, const int* __restrict__ seg_start,
    float* __restrict__ ex, float* __restrict__ seg_recip)
{
  __shared__ float red[256];
  const int g = blockIdx.x;
  const int t = threadIdx.x;
  const int begin = seg_start[g];
  const int end   = seg_start[g + 1];

  float m = -INFINITY;
  for (int i = begin + t; i < end; i += 256) m = fmaxf(m, scores[i]);
  red[t] = m;
  __syncthreads();
  for (int s = 128; s > 0; s >>= 1) {
    if (t < s) red[t] = fmaxf(red[t], red[t + s]);
    __syncthreads();
  }
  m = red[0];
  if (!(m > -INFINITY)) m = 0.0f;   // empty segment guard (matches isfinite)
  __syncthreads();

  float sum = 0.0f;
  for (int i = begin + t; i < end; i += 256) {
    float e = __expf(scores[i] - m);
    ex[i] = e;
    sum += e;
  }
  red[t] = sum;
  __syncthreads();
  for (int s = 128; s > 0; s >>= 1) {
    if (t < s) red[t] += red[t + s];
    __syncthreads();
  }
  if (t == 0) seg_recip[g] = 1.0f / (red[0] + 1e-16f);
}

// ---------------------------------------------------------------------------
// Kernel 4: out[g][d] = sum_{n in segment g} x[n][d] * ex[n] * recip[g]
// One block per segment, thread-per-column; alpha staged in LDS chunks.
// ---------------------------------------------------------------------------
__global__ __launch_bounds__(256) void pool_kernel(
    const float* __restrict__ x, const float* __restrict__ ex,
    const float* __restrict__ seg_recip, const int* __restrict__ seg_start,
    float* __restrict__ out)
{
  __shared__ float s_alpha[256];
  const int g = blockIdx.x;
  const int d = threadIdx.x;
  const int begin = seg_start[g];
  const int end   = seg_start[g + 1];
  const float inv = seg_recip[g];

  float acc = 0.0f;
  for (int base = begin; base < end; base += 256) {
    int cnt = min(256, end - base);
    __syncthreads();
    if (d < cnt) s_alpha[d] = ex[base + d] * inv;
    __syncthreads();
#pragma unroll 4
    for (int j = 0; j < cnt; ++j)
      acc += x[(size_t)(base + j) * D_DIM + d] * s_alpha[j];
  }
  out[(size_t)g * D_DIM + d] = acc;   // empty segment -> 0, matches reference
}

// ---------------------------------------------------------------------------
extern "C" void kernel_launch(void* const* d_in, const int* in_sizes, int n_in,
                              void* d_out, int out_size, void* d_ws, size_t ws_size,
                              hipStream_t stream)
{
  const float*     x     = (const float*)d_in[0];
  const long long* batch = (const long long*)d_in[1];   // int64 per reference
  const float*     W1    = (const float*)d_in[2];
  const float*     b1    = (const float*)d_in[3];
  const float*     W2    = (const float*)d_in[4];
  const float*     b2    = (const float*)d_in[5];
  float*           out   = (float*)d_out;
  const int n = in_sizes[0] / D_DIM;

  // Workspace layout
  char* ws = (char*)d_ws;
  float* scores    = (float*)ws;  ws += (size_t)n * sizeof(float);
  float* ex        = (float*)ws;  ws += (size_t)n * sizeof(float);
  float* seg_recip = (float*)ws;  ws += (size_t)NSEG * sizeof(float);
  int*   seg_start = (int*)ws;    // NSEG + 1 ints

  seg_bounds_kernel<<<(n + 255) / 256, 256, 0, stream>>>(batch, seg_start, n);
  attn_scores_kernel<<<(n + ROWS_PER_BLOCK - 1) / ROWS_PER_BLOCK, 256, 0, stream>>>(
      x, W1, b1, W2, b2, scores, n);
  seg_stats_kernel<<<NSEG, 256, 0, stream>>>(scores, seg_start, ex, seg_recip);
  pool_kernel<<<NSEG, 256, 0, stream>>>(x, ex, seg_recip, seg_start, out);
}
